// PyRNN_20538533609995
// MI455X (gfx1250) — matrix operand order repaired and checked by reading, not compiled, not run on hardware
//
#include <hip/hip_runtime.h>

// ---------------------------------------------------------------------------
// Multi-layer tanh RNN (B=32, L=512, H=1024, 4 layers) for MI455X / gfx1250.
// Persistent bf16-WMMA kernel: 4 layers x 16 WGs, weights LDS-resident,
// wavefront pipelining across (layer, t) with device-scope flags.
// ---------------------------------------------------------------------------

#define BB   32      // batch
#define LL   512     // sequence length
#define HH   1024    // hidden / input dim
#define NL   4       // layers
#define NWG  16      // workgroups per layer
#define ROWS 64      // output rows (columns of h) per workgroup = HH/NWG
#define KTOT 2048    // concatenated K: [inp | h_prev]
#define LDSK 2056    // padded LDS K-stride (row stride 4112B -> bank-spread)

typedef __attribute__((ext_vector_type(16))) __bf16 v16bf;
typedef __attribute__((ext_vector_type(8)))  __bf16 v8bf;
typedef __attribute__((ext_vector_type(8)))  float  v8f;

__device__ __forceinline__ __bf16 f2bf(float f) {
  unsigned u = __float_as_uint(f);
  u += 0x7FFFu + ((u >> 16) & 1u);      // round-to-nearest-even
  unsigned short h = (unsigned short)(u >> 16);
  return __builtin_bit_cast(__bf16, h);
}

// Load 16 bf16 as two 16B chunks separated by `d` elements (A: d=16, B: d=16).
__device__ __forceinline__ v16bf ldpair(const __bf16* p, int d) {
  v8bf lo = *(const v8bf*)(p);
  v8bf hi = *(const v8bf*)(p + d);
  return __builtin_shufflevector(lo, hi, 0,1,2,3,4,5,6,7,8,9,10,11,12,13,14,15);
}

__device__ __forceinline__ void wait_flag(int* f, int target) {
  if (threadIdx.x == 0) {
    while (__hip_atomic_load(f, __ATOMIC_ACQUIRE, __HIP_MEMORY_SCOPE_AGENT) < target)
      __builtin_amdgcn_s_sleep(4);
  }
  __syncthreads();
}

// --------------------------- prep kernels ----------------------------------

// x [B][L][H] fp32 -> xbf [L][B][H] bf16 (time-major for per-step contiguity)
__global__ __launch_bounds__(256) void PyRNN_conv_x(const float* __restrict__ x,
                                                    __bf16* __restrict__ xbf) {
  unsigned i = blockIdx.x * 256u + threadIdx.x;   // dest index, LL*BB*HH total
  unsigned k = i & (HH - 1);
  unsigned b = (i >> 10) & (BB - 1);
  unsigned t = i >> 15;
  xbf[i] = f2bf(x[(((size_t)b * LL + t) << 10) + k]);
}

// WI,WH [NL][H][H] fp32 -> wpack [NL][H][KTOT] bf16, row n: [WI[n,:] | WH[n,:]]
__global__ __launch_bounds__(256) void PyRNN_pack_w(const float* __restrict__ WI,
                                                    const float* __restrict__ WH,
                                                    __bf16* __restrict__ wpack) {
  unsigned i = blockIdx.x * 256u + threadIdx.x;   // NL*HH*KTOT total
  unsigned k = i & (KTOT - 1);
  unsigned n = (i >> 11) & (HH - 1);
  unsigned l = i >> 21;
  float v = (k < HH) ? WI[((size_t)l * HH + n) * HH + k]
                     : WH[((size_t)l * HH + n) * HH + (k - HH)];
  wpack[i] = f2bf(v);
}

// Fused bias, h0 -> hbuf slot 0, flag reset (every launch: graph-replay safe)
__global__ __launch_bounds__(256) void PyRNN_init(const float* __restrict__ h0,
                                                  const float* __restrict__ BI,
                                                  const float* __restrict__ BH,
                                                  __bf16* __restrict__ hbuf,
                                                  float* __restrict__ biasC,
                                                  int* __restrict__ flags) {
  unsigned i = blockIdx.x * 256u + threadIdx.x;
  if (i < NL * BB * HH) {
    unsigned l = i / (BB * HH);
    unsigned r = i - l * (BB * HH);
    hbuf[(size_t)l * (LL + 1) * BB * HH + r] = f2bf(h0[i]);
  }
  if (i < NL * HH) biasC[i] = BI[i] + BH[i];
  if (i < NL * LL) flags[i] = 0;
}

// --------------------------- main persistent kernel ------------------------

__global__ __launch_bounds__(256, 1) void PyRNN_persistent(
    const __bf16* __restrict__ xbf,    // [LL][BB][HH]
    const __bf16* __restrict__ wpack,  // [NL][HH][KTOT]
    const float*  __restrict__ biasC,  // [NL][HH]
    __bf16* __restrict__ hbuf,         // [NL][LL+1][BB][HH]
    int*    __restrict__ flags,        // [NL][LL]
    float*  __restrict__ out,          // [BB][LL][HH]
    float*  __restrict__ hfinal) {     // [NL][BB][HH]
  extern __shared__ __bf16 lds[];      // ROWS x LDSK bf16 (263,168 B)

  const int l    = blockIdx.x >> 4;
  const int g    = blockIdx.x & 15;
  const int tid  = threadIdx.x;
  const int lane = tid & 31;
  const int wave = tid >> 5;
  const int half = lane >> 4;          // half-wave select
  const int l16  = lane & 15;
  const int mt   = wave & 1;           // M tile (batch rows 0-15 / 16-31)
  const int nt   = wave >> 1;          // N tile within 64-col slice
  const int n0   = g * ROWS;

  // ---- stage this WG's 64x2048 bf16 weight slice into LDS (once) ----
  {
    const __bf16* src = wpack + ((size_t)l * HH + n0) * KTOT;
    for (int c = tid; c < ROWS * (KTOT / 8); c += 256) {   // 16B chunks
      int row = c >> 8;                // KTOT/8 = 256 chunks per row
      int col = (c & 255) << 3;
      *(v8bf*)(lds + row * LDSK + col) =
          *(const v8bf*)(src + (size_t)row * KTOT + col);
    }
  }
  __syncthreads();

  const int m     = mt * 16 + l16;                 // A-matrix row (batch)
  const int nLoc  = nt * 16 + l16;                 // B-matrix column (local)
  const int nGlob = n0 + nLoc;
  const int aoff  = half ? 8 : 0;                  // A K-offset per half-wave
  const int boff  = half ? 8 : 0;                  // B K-offset per half-wave
  const float bias = biasC[l * HH + nGlob];
  int* flg     = flags + l * LL;
  int* flgPrev = flags + (l - 1) * LL;             // deref'd only when l>0

  for (int t = 0; t < LL; ++t) {
    if (l == 0 && t + 1 < LL)
      __builtin_prefetch(xbf + ((size_t)(t + 1) * BB + m) * HH, 0, 1);

    // wait for producers (monotonic flags; t==0 inputs set by init kernel)
    if (l > 0) wait_flag(&flgPrev[t], NWG);
    if (t > 0) wait_flag(&flg[t - 1], NWG);

    const __bf16* inpA = (l == 0)
        ? (xbf + (size_t)t * BB * HH)
        : (hbuf + ((size_t)(l - 1) * (LL + 1) + (t + 1)) * BB * HH);
    const __bf16* hprA = hbuf + ((size_t)l * (LL + 1) + t) * BB * HH;

    v8f acc = {};
    {   // K = 0..1023 : input-path GEMM (weights at LDS K base 0)
      const __bf16* arow = inpA + (size_t)m * HH + aoff;
      const __bf16* wrow = lds + (size_t)nLoc * LDSK + boff;
      #pragma unroll 4
      for (int k = 0; k < HH; k += 32) {
        v16bf A  = ldpair(arow + k, 16);
        v16bf Bf = ldpair(wrow + k, 16);
        acc = __builtin_amdgcn_wmma_f32_16x16x32_bf16(
            false, A, false, Bf, (short)0, acc, false, false);
      }
    }
    {   // K = 1024..2047 : recurrent-path GEMM (weights at LDS K base HH)
      const __bf16* arow = hprA + (size_t)m * HH + aoff;
      const __bf16* wrow = lds + (size_t)nLoc * LDSK + HH + boff;
      #pragma unroll 4
      for (int k = 0; k < HH; k += 32) {
        v16bf A  = ldpair(arow + k, 16);
        v16bf Bf = ldpair(wrow + k, 16);
        acc = __builtin_amdgcn_wmma_f32_16x16x32_bf16(
            false, A, false, Bf, (short)0, acc, false, false);
      }
    }

    // epilogue: bias + tanh, publish bf16 h, fp32 outputs
    float vals[8];
    #pragma unroll
    for (int i = 0; i < 8; ++i) vals[i] = tanhf(acc[i] + bias);

    __bf16* hOut = hbuf + ((size_t)l * (LL + 1) + (t + 1)) * BB * HH;
    #pragma unroll
    for (int i = 0; i < 8; ++i) {
      int mRow = mt * 16 + i + half * 8;           // C/D VGPR i -> M row
      hOut[(size_t)mRow * HH + nGlob] = f2bf(vals[i]);
      if (l == NL - 1) out[((size_t)mRow * LL + t) * HH + nGlob] = vals[i];
      if (t == LL - 1) hfinal[((size_t)l * BB + mRow) * HH + nGlob] = vals[i];
    }

    __threadfence();                                // stores -> agent scope
    __syncthreads();
    if (tid == 0)
      __hip_atomic_fetch_add(&flg[t], 1, __ATOMIC_RELEASE,
                             __HIP_MEMORY_SCOPE_AGENT);
  }
}

// --------------------------- launch ----------------------------------------

extern "C" void kernel_launch(void* const* d_in, const int* in_sizes, int n_in,
                              void* d_out, int out_size, void* d_ws, size_t ws_size,
                              hipStream_t stream) {
  const float* x  = (const float*)d_in[0];
  const float* h0 = (const float*)d_in[1];
  const float* WI = (const float*)d_in[2];
  const float* BI = (const float*)d_in[3];
  const float* WH = (const float*)d_in[4];
  const float* BH = (const float*)d_in[5];

  float* out    = (float*)d_out;                       // [BB][LL][HH]
  float* hfinal = out + (size_t)BB * LL * HH;          // [NL][BB][HH]

  char* ws = (char*)d_ws;
  __bf16* xbf   = (__bf16*)ws;  ws += (size_t)LL * BB * HH * 2;          // 33.6 MB
  __bf16* wpack = (__bf16*)ws;  ws += (size_t)NL * HH * KTOT * 2;        // 16.8 MB
  __bf16* hbuf  = (__bf16*)ws;  ws += (size_t)NL * (LL + 1) * BB * HH * 2; // 134.5 MB
  float*  biasC = (float*)ws;   ws += (size_t)NL * HH * 4;
  int*    flags = (int*)ws;

  PyRNN_conv_x<<<(LL * BB * HH) / 256, 256, 0, stream>>>(x, xbf);
  PyRNN_pack_w<<<(NL * HH * KTOT) / 256, 256, 0, stream>>>(WI, WH, wpack);
  PyRNN_init<<<(NL * BB * HH) / 256, 256, 0, stream>>>(h0, BI, BH, hbuf, biasC, flags);

  const size_t ldsBytes = (size_t)ROWS * LDSK * sizeof(__bf16);          // 263,168 B
  PyRNN_persistent<<<NL * NWG, 256, ldsBytes, stream>>>(
      xbf, wpack, biasC, hbuf, flags, out, hfinal);
}
